// MultiHeadedAttention_8761733284297
// MI455X (gfx1250) — matrix operand order repaired and checked by reading, not compiled, not run on hardware
//
#include <hip/hip_runtime.h>
#include <hip/hip_bf16.h>

// ---------------- MHA constants (match reference) ----------------
#define BATCH   4
#define SEQ     2048
#define DMODEL  512
#define HEADS   8
#define DHEAD   64
#define MROWS   (BATCH * SEQ)          // 8192

// ---------------- vector types ----------------
typedef __attribute__((ext_vector_type(16))) __bf16         v16bf;
typedef __attribute__((ext_vector_type(8)))  __bf16         v8bf;
typedef __attribute__((ext_vector_type(8)))  float          v8f;
typedef __attribute__((ext_vector_type(4)))  unsigned int   v4u;
typedef __attribute__((ext_vector_type(8)))  unsigned short v8us;

// ---------------- helpers ----------------
static __device__ __forceinline__ unsigned short f2bf(float f) {
    unsigned int u = __float_as_uint(f);
    u += 0x7FFFu + ((u >> 16) & 1u);   // round-to-nearest-even
    return (unsigned short)(u >> 16);
}
static __device__ __forceinline__ unsigned int pack2bf(float a, float b) {
    return (unsigned int)f2bf(a) | ((unsigned int)f2bf(b) << 16);
}
static __device__ __forceinline__ v8bf load8(const unsigned short* p) {
    return __builtin_bit_cast(v8bf, *(const v4u*)p);   // 16B load
}
static __device__ __forceinline__ v16bf cat16(v8bf a, v8bf b) {
    return __builtin_shufflevector(a, b, 0,1,2,3,4,5,6,7,8,9,10,11,12,13,14,15);
}
static __device__ __forceinline__ v16bf lds16(const unsigned short* p) {
    return cat16(load8(p), load8(p + 8));              // contiguous 16 bf16
}
static __device__ __forceinline__ v8f bf_wmma(v16bf a, v16bf b, v8f c) {
    return __builtin_amdgcn_wmma_f32_16x16x32_bf16(false, a, false, b,
                                                   (short)0, c, false, false);
}
// generic -> LDS byte address (for async-load VDST operand)
static __device__ __forceinline__ unsigned lds_addr(const void* p) {
    return (unsigned)(uintptr_t)(__attribute__((address_space(3))) const void*)p;
}

// =================================================================
// GEMM:  C[m][n] = sum_k X[m][k] * W[n][k] + bias[n]
//   K = N = DMODEL = 512.  Block tile 128x64, 8 waves.
//   Wave w: rows (w&3)*32, cols (w>>2)*32 -> 32x32 output
//   (4 accumulators, 2 A-frags x 2 B-frags, 4 WMMAs per K-step).
//   MODE 0: output bf16 written to [B,H,S,Dh]   (Q/K/V projections)
//   MODE 1: output fp32 row-major [M, DMODEL]   (output projection)
// =================================================================
template <int MODE>
__global__ __launch_bounds__(256)
void mha_proj_gemm(const float* __restrict__ X, const float* __restrict__ W,
                   const float* __restrict__ bias, void* __restrict__ out) {
    __shared__ unsigned short Xs[128 * 32];  // 8 KB  (bf16 A tile, [m][k])
    __shared__ unsigned short Ws[64 * 32];   // 4 KB  (bf16 B tile, [n][k])

    const int tid  = threadIdx.x;
    const int w    = tid >> 5;
    const int lane = tid & 31;
    const int lm   = lane & 15;
    const int kh   = lane >> 4;

    const int m0 = blockIdx.y * 128;
    const int n0 = blockIdx.x * 64;
    const int mt = (w & 3) * 32;
    const int nt = (w >> 2) * 32;

    const int xr = tid >> 1;        // 0..127: X tile row staged by this thread
    const int xc = (tid & 1) * 16;  // 0/16:   k-group of 16
    const int wr = tid >> 2;        // 0..63:  W tile row
    const int wc = (tid & 3) * 8;   // k-group of 8

    v8f acc00 = {}, acc01 = {}, acc10 = {}, acc11 = {};

    for (int kk = 0; kk < DMODEL; kk += 32) {
        __syncthreads();
        // ---- stage X tile (fp32 -> bf16) ----
        {
            const float* src = X + (size_t)(m0 + xr) * DMODEL + kk + xc;
            unsigned int* dst = (unsigned int*)&Xs[xr * 32 + xc];
#pragma unroll
            for (int i = 0; i < 8; ++i)
                dst[i] = pack2bf(src[2 * i], src[2 * i + 1]);
        }
        // ---- stage W tile (fp32 -> bf16) ----
        {
            const float* src = W + (size_t)(n0 + wr) * DMODEL + kk + wc;
            unsigned int* dst = (unsigned int*)&Ws[wr * 32 + wc];
#pragma unroll
            for (int i = 0; i < 4; ++i)
                dst[i] = pack2bf(src[2 * i], src[2 * i + 1]);
        }
        if (kk + 32 < DMODEL)
            __builtin_prefetch(X + (size_t)(m0 + xr) * DMODEL + kk + 32 + xc, 0, 3);
        __syncthreads();

        // ---- fragments (contiguous k = 16*kh + s in both A and B) ----
        const unsigned short* ar0 = &Xs[(mt + lm) * 32];
        const unsigned short* ar1 = &Xs[(mt + 16 + lm) * 32];
        v16bf a0 = lds16(ar0 + 16 * kh);
        v16bf a1 = lds16(ar1 + 16 * kh);
        v16bf b0 = lds16(&Ws[(nt + lm) * 32 + 16 * kh]);
        v16bf b1 = lds16(&Ws[(nt + 16 + lm) * 32 + 16 * kh]);

        acc00 = bf_wmma(a0, b0, acc00);
        acc01 = bf_wmma(a0, b1, acc01);
        acc10 = bf_wmma(a1, b0, acc10);
        acc11 = bf_wmma(a1, b1, acc11);
    }

    // ---- epilogue: bias + store ----
    const int nA = n0 + nt + lm;
    const int nB = nA + 16;
    const float bA = bias[nA];
    const float bB = bias[nB];
#pragma unroll
    for (int r = 0; r < 8; ++r) {
        const int mLo = m0 + mt + r + 8 * kh;       // C layout: row = r + 8*(lane>>4)
        const int mHi = mLo + 16;
        const float v00 = acc00[r] + bA;
        const float v01 = acc01[r] + bB;
        const float v10 = acc10[r] + bA;
        const float v11 = acc11[r] + bB;
        if (MODE == 0) {
            // scatter into [B, H, S, Dh] bf16
            unsigned short* o = (unsigned short*)out;
            const int bLo = mLo >> 11, sLo = mLo & (SEQ - 1);
            const int bHi = mHi >> 11, sHi = mHi & (SEQ - 1);
            const int hA = nA >> 6, dA = nA & 63;
            const int hB = nB >> 6, dB = nB & 63;
            o[(((size_t)bLo * HEADS + hA) * SEQ + sLo) * DHEAD + dA] = f2bf(v00);
            o[(((size_t)bLo * HEADS + hB) * SEQ + sLo) * DHEAD + dB] = f2bf(v01);
            o[(((size_t)bHi * HEADS + hA) * SEQ + sHi) * DHEAD + dA] = f2bf(v10);
            o[(((size_t)bHi * HEADS + hB) * SEQ + sHi) * DHEAD + dB] = f2bf(v11);
        } else {
            float* o = (float*)out;
            o[(size_t)mLo * DMODEL + nA] = v00;
            o[(size_t)mLo * DMODEL + nB] = v01;
            o[(size_t)mHi * DMODEL + nA] = v10;
            o[(size_t)mHi * DMODEL + nB] = v11;
        }
    }
}

// =================================================================
// Flash attention over one (b, h): block = 8 waves x 16 queries = 128 q.
// Q/K/V bf16 in [B,H,S,Dh]; output fp32 [B,S,H*Dh] (heads re-interleaved).
// K chunk staged with CDNA5 async load-to-LDS (ASYNCcnt path).
// =================================================================
__global__ __launch_bounds__(256)
void mha_flash_attn(const unsigned short* __restrict__ Q,
                    const unsigned short* __restrict__ K,
                    const unsigned short* __restrict__ V,
                    float* __restrict__ attnOut) {
    __shared__ unsigned short Ks[32 * 64];      // 4 KB  [key][d]
    __shared__ unsigned short Vst[64 * 32];     // 4 KB  [d][key]  (transposed)
    __shared__ unsigned short Ps[8][16 * 32];   // 8 KB  per-wave P transpose

    const int tid  = threadIdx.x;
    const int w    = tid >> 5;
    const int lane = tid & 31;
    const int lm   = lane & 15;
    const int kh   = lane >> 4;

    const int b = blockIdx.z, h = blockIdx.y;
    const int qbase = blockIdx.x * 128 + w * 16;
    const size_t headOff = ((size_t)b * HEADS + h) * SEQ * DHEAD;

    // ---- Q A-fragments, kept in registers for the whole key loop ----
    const unsigned short* qrow = Q + headOff + (size_t)(qbase + lm) * DHEAD;
    v16bf qf0 = cat16(load8(qrow + 16 * kh),      load8(qrow + 8 + 16 * kh)); // d 0..31
    v16bf qf1 = cat16(load8(qrow + 32 + 16 * kh), load8(qrow + 40 + 16 * kh)); // d 32..63

    float mrow[8], lrow[8];
    v8f o0 = {}, o1 = {}, o2 = {}, o3 = {};
#pragma unroll
    for (int r = 0; r < 8; ++r) { mrow[r] = -1e30f; lrow[r] = 0.0f; }

    const int lkey = tid >> 3;       // 0..31: key row this thread stages
    const int ld0  = (tid & 7) * 8;  // d-group of 8
    const unsigned ldsK = lds_addr(&Ks[lkey * 64 + ld0]);

    for (int kc = 0; kc < SEQ; kc += 32) {
        __syncthreads();
        // ---- stage K chunk [32][64] via async load-to-LDS (no VGPR bounce) ----
        {
            const unsigned short* gK = K + headOff + (size_t)(kc + lkey) * DHEAD + ld0;
            asm volatile("global_load_async_to_lds_b128 %0, %1, off"
                         :: "v"(ldsK), "v"(gK) : "memory");
        }
        // ---- stage V chunk transposed -> Vst[d][key] (needs VGPR transpose) ----
        {
            v4u vv = *(const v4u*)(V + headOff + (size_t)(kc + lkey) * DHEAD + ld0);
            v8us ve = __builtin_bit_cast(v8us, vv);
#pragma unroll
            for (int i = 0; i < 8; ++i)
                Vst[(ld0 + i) * 32 + lkey] = ve[i];
        }
        asm volatile("s_wait_asynccnt 0x0" ::: "memory");
        __syncthreads();

        // ---- scores: s = Q . K^T  (two 16x16 tiles over 32 keys) ----
        const unsigned short* kr0 = &Ks[lm * 64];
        const unsigned short* kr1 = &Ks[(16 + lm) * 64];
        v8f s0 = {}, s1 = {};
        s0 = bf_wmma(qf0, lds16(kr0 + 16 * kh), s0);
        s0 = bf_wmma(qf1, lds16(kr0 + 32 + 16 * kh), s0);
        s1 = bf_wmma(qf0, lds16(kr1 + 16 * kh), s1);
        s1 = bf_wmma(qf1, lds16(kr1 + 32 + 16 * kh), s1);

        // ---- online softmax (rows live across 16-lane halves) ----
        const float scale = 0.125f;  // 1/sqrt(64)
#pragma unroll
        for (int r = 0; r < 8; ++r) {
            float a  = s0[r] * scale;
            float c  = s1[r] * scale;
            float mx = fmaxf(a, c);
            mx = fmaxf(mx, __shfl_xor(mx, 1, 32));
            mx = fmaxf(mx, __shfl_xor(mx, 2, 32));
            mx = fmaxf(mx, __shfl_xor(mx, 4, 32));
            mx = fmaxf(mx, __shfl_xor(mx, 8, 32));
            float mnew = fmaxf(mrow[r], mx);
            float al   = __expf(mrow[r] - mnew);
            float p0   = __expf(a - mnew);
            float p1   = __expf(c - mnew);
            float rs   = p0 + p1;
            rs += __shfl_xor(rs, 1, 32);
            rs += __shfl_xor(rs, 2, 32);
            rs += __shfl_xor(rs, 4, 32);
            rs += __shfl_xor(rs, 8, 32);
            lrow[r] = lrow[r] * al + rs;
            mrow[r] = mnew;
            // C-layout -> LDS for A-layout re-read (row = r+8*kh, col = key)
            Ps[w][(r + 8 * kh) * 32 + lm]      = f2bf(p0);
            Ps[w][(r + 8 * kh) * 32 + 16 + lm] = f2bf(p1);
            o0[r] *= al; o1[r] *= al; o2[r] *= al; o3[r] *= al;
        }
        // same-wave LDS RAW: DS ops are in-order, fence the compiler + counter
        asm volatile("s_wait_dscnt 0x0" ::: "memory");

        // ---- P A-fragment (16 queries x 32 keys) ----
        const unsigned short* prow = &Ps[w][lm * 32];
    v16bf pf = cat16(load8(prow + 16 * kh), load8(prow + 8 + 16 * kh));

        // ---- O += P . V  (four 16x16 d-tiles, one WMMA each) ----
        o0 = bf_wmma(pf, lds16(&Vst[(lm)      * 32 + 16 * kh]), o0);
        o1 = bf_wmma(pf, lds16(&Vst[(16 + lm) * 32 + 16 * kh]), o1);
        o2 = bf_wmma(pf, lds16(&Vst[(32 + lm) * 32 + 16 * kh]), o2);
        o3 = bf_wmma(pf, lds16(&Vst[(48 + lm) * 32 + 16 * kh]), o3);
    }

    // ---- normalize and write [B, S, H*Dh] fp32 ----
#pragma unroll
    for (int r = 0; r < 8; ++r) {
        const float inv = 1.0f / lrow[r];
        const int q = qbase + r + 8 * kh;
        float* orow = attnOut + ((size_t)b * SEQ + q) * DMODEL + h * DHEAD;
        orow[lm]      = o0[r] * inv;
        orow[16 + lm] = o1[r] * inv;
        orow[32 + lm] = o2[r] * inv;
        orow[48 + lm] = o3[r] * inv;
    }
}

// =================================================================
// Launch: 3x QKV projection GEMMs -> flash attention -> output GEMM
// =================================================================
extern "C" void kernel_launch(void* const* d_in, const int* in_sizes, int n_in,
                              void* d_out, int out_size, void* d_ws, size_t ws_size,
                              hipStream_t stream) {
    (void)in_sizes; (void)n_in; (void)out_size; (void)ws_size;
    const float* X  = (const float*)d_in[0];
    const float* Wq = (const float*)d_in[1];
    const float* bq = (const float*)d_in[2];
    const float* Wk = (const float*)d_in[3];
    const float* bk = (const float*)d_in[4];
    const float* Wv = (const float*)d_in[5];
    const float* bv = (const float*)d_in[6];
    const float* Wo = (const float*)d_in[7];
    const float* bo = (const float*)d_in[8];

    // workspace layout: Q,K,V bf16 (8 MB each) + attn fp32 (16 MB)
    const size_t headElems = (size_t)BATCH * HEADS * SEQ * DHEAD;  // 4M
    unsigned short* Qb = (unsigned short*)d_ws;
    unsigned short* Kb = Qb + headElems;
    unsigned short* Vb = Kb + headElems;
    float* attnB = (float*)(Vb + headElems);

    dim3 gGemm(DMODEL / 64, MROWS / 128);  // (8, 64)
    dim3 blk(256);

    mha_proj_gemm<0><<<gGemm, blk, 0, stream>>>(X, Wq, bq, (void*)Qb);
    mha_proj_gemm<0><<<gGemm, blk, 0, stream>>>(X, Wk, bk, (void*)Kb);
    mha_proj_gemm<0><<<gGemm, blk, 0, stream>>>(X, Wv, bv, (void*)Vb);

    dim3 gAttn(SEQ / 128, HEADS, BATCH);   // (16, 8, 4)
    mha_flash_attn<<<gAttn, blk, 0, stream>>>(Qb, Kb, Vb, attnB);

    mha_proj_gemm<1><<<gGemm, blk, 0, stream>>>(attnB, Wo, bo, d_out);
}
